// TritonMSARowAttention_63797444215477
// MI455X (gfx1250) — matrix operand order repaired and checked by reading, not compiled, not run on hardware
//
#include <hip/hip_runtime.h>

// ---------------------------------------------------------------------------
// MSA row attention for MI455X (gfx1250, wave32, WMMA f32_16x16x32_f16).
// msa/weights pre-swizzled into exact WMMA fragment layouts -> LDS-free GEMMs
// (pure global_load_b128 + v_wmma); flash attention with double-buffered
// global_load_async_to_lds_b128 K/V staging (partial ASYNCcnt waits) and a
// per-wave LDS P relayout. Softmax runs in base-2 (log2e folded into q).
// ---------------------------------------------------------------------------

typedef _Float16 h16;
typedef __attribute__((ext_vector_type(16))) _Float16 v16h;
typedef __attribute__((ext_vector_type(8)))  float    v8f;

#define SEQ   128
#define LEN   384
#define DIM   256
#define HEADS 8
#define HDIM  32
#define MTOT  (SEQ * LEN)            // 49152
#define SCALE 0.17677669529663687f   // 1/sqrt(32)
#define QSC   (SCALE * 1.4426950408889634f)   // base-2 softmax scale

union Frag { v16h v; uint4 u[2]; };
union H8   { uint4 u; h16 h[8]; };

// Native v_exp_f32 (base-2 exponential).
__device__ __forceinline__ float exp2_fast(float x) {
  return __builtin_amdgcn_exp2f(x);
}

__device__ __forceinline__ v8f wmma16(v16h a, v16h b, v8f c) {
  return __builtin_amdgcn_wmma_f32_16x16x32_f16(
      false, a, false, b, (short)0, c, false, false);
}

// 32B contiguous global->LDS copy on the gfx1250 async path (ASYNCcnt).
// INST_OFFSET applies to BOTH the LDS and global address, so offset:16
// gives the second b128 of the run with the same address VGPRs.
__device__ __forceinline__ void cp_async32(const h16* g, const h16* l) {
  unsigned la = (unsigned)(size_t)l;   // low 32 bits of generic LDS addr
  asm volatile("global_load_async_to_lds_b128 %0, %1, off\n\t"
               "global_load_async_to_lds_b128 %0, %1, off offset:16"
               :: "v"(la), "v"(g) : "memory");
}
// Async loads complete in order: waiting ASYNCcnt<=2 guarantees everything
// older than this wave's 2 newest issues has landed in LDS.
__device__ __forceinline__ void wait_async0() {
  asm volatile("s_wait_asynccnt 0" ::: "memory");
}
__device__ __forceinline__ void wait_async2() {
  asm volatile("s_wait_asynccnt 2" ::: "memory");
}

// Row reductions across the 16-lane half that owns one C-tile row (wave32).
__device__ __forceinline__ float redmax16(float x) {
  x = fmaxf(x, __shfl_xor(x, 1, 32));
  x = fmaxf(x, __shfl_xor(x, 2, 32));
  x = fmaxf(x, __shfl_xor(x, 4, 32));
  x = fmaxf(x, __shfl_xor(x, 8, 32));
  return x;
}
__device__ __forceinline__ float redsum16(float x) {
  x += __shfl_xor(x, 1, 32);
  x += __shfl_xor(x, 2, 32);
  x += __shfl_xor(x, 4, 32);
  x += __shfl_xor(x, 8, 32);
  return x;
}

// ---------------------------------------------------------------------------
// Kernel 1: pack 4 weights f32->f16 into B-fragment swizzle:
// wsw[w][kb(8)][nt(16)][lane(32)][e(16)], value = W[n][k],
// lane = (n&15) + 16*((k>>3)&1), e = (k&7) + 8*((k>>4)&1).
// ---------------------------------------------------------------------------
__global__ __launch_bounds__(256) void k_pack_w(
    const float* __restrict__ Wq, const float* __restrict__ Wk,
    const float* __restrict__ Wv, const float* __restrict__ Wo,
    h16* __restrict__ wsw) {
  int idx = blockIdx.x * 256 + threadIdx.x;    // 4*65536
  int w = idx >> 16, r = idx & 65535;
  int k = r >> 8, n = r & 255;
  const float* W = (w == 0) ? Wq : (w == 1) ? Wk : (w == 2) ? Wv : Wo;
  size_t dst = ((size_t)((w * 8 + (k >> 5)) * 16 + (n >> 4))) * 512 +
               ((n & 15) + 16 * ((k >> 3) & 1)) * 16 + ((k & 7) + 8 * ((k >> 4) & 1));
  wsw[dst] = (h16)W[n * 256 + k];
}

// ---------------------------------------------------------------------------
// Kernel 2: msa f32 -> f16 A-fragment swizzle: asw[mtile(3072)][kb(8)][lane][e]
// A 16x32 frag: lane = (m&15)+16*((k>>3)&1), halves e: K = e+8*(e>>3)+8*(lane>>4).
// Each aligned 8-wide k-run maps to one contiguous 16B run.
// ---------------------------------------------------------------------------
__global__ __launch_bounds__(256) void k_swz_a(
    const float* __restrict__ msa, h16* __restrict__ asw) {
  int idx = blockIdx.x * 256 + threadIdx.x;    // 49152*32 runs of 8
  int m = idx >> 5, run = idx & 31;
  const float4* src = (const float4*)(msa + (size_t)m * 256 + run * 8);
  float4 f0 = src[0], f1 = src[1];
  H8 o;
  o.h[0] = (h16)f0.x; o.h[1] = (h16)f0.y; o.h[2] = (h16)f0.z; o.h[3] = (h16)f0.w;
  o.h[4] = (h16)f1.x; o.h[5] = (h16)f1.y; o.h[6] = (h16)f1.z; o.h[7] = (h16)f1.w;
  int kb = run >> 2, g = run & 1, e0 = 8 * ((run >> 1) & 1);
  size_t dst = ((size_t)(m >> 4) * 8 + kb) * 512 + ((m & 15) + 16 * g) * 16 + e0;
  *(uint4*)(asw + dst) = o.u;
}

// ---------------------------------------------------------------------------
// Kernel 3: QKV projection, LDS-free. 8 waves/block, wave = 32x64 C tile.
// Per k-step: 2 A frags + 4 B frags (wave-coalesced b128 bursts) + 8 WMMAs.
// Outputs written directly into attention-ready swizzled layouts:
//   qsw[s][h][ltile(24)][lane][e]  (A-frag of q rows, QSC folded in)
//   ksw[s][h][mtile(24)][lane][e]  (B-frag tiles of k^T: 32d x 16m)
//   vsw[s][h][kblk(12)][dt(2)][lane][e] (B-frag tiles of v: 32m x 16d)
// ---------------------------------------------------------------------------
__global__ __launch_bounds__(256) void k_qkv(
    const h16* __restrict__ asw, const h16* __restrict__ wsw,
    h16* __restrict__ qsw, h16* __restrict__ ksw, h16* __restrict__ vsw) {
  const int mode = blockIdx.z, bx = blockIdx.x;
  const int nBase = blockIdx.y * 128;
  const int t = threadIdx.x, lane = t & 31, wid = t >> 5;
  const int mi2 = wid >> 1, ni2 = wid & 1;
  const int g = lane >> 4, ln = lane & 15;
  const int mt0 = bx * 8 + mi2 * 2;
  const int nt0 = nBase / 16 + ni2 * 4;

  v8f zero = {0.f, 0.f, 0.f, 0.f, 0.f, 0.f, 0.f, 0.f};
  v8f c[2][4];
#pragma unroll
  for (int i = 0; i < 2; ++i)
#pragma unroll
    for (int j = 0; j < 4; ++j) c[i][j] = zero;

  const h16* A0 = asw + (size_t)mt0 * 8 * 512 + lane * 16;
  const h16* A1 = A0 + 8 * 512;
  const h16* B0 = wsw + ((size_t)mode * 8 * 16 + nt0) * 512 + lane * 16;
#pragma unroll
  for (int kb = 0; kb < 8; ++kb) {
    v16h a0 = *(const v16h*)(A0 + kb * 512);
    v16h a1 = *(const v16h*)(A1 + kb * 512);
#pragma unroll
    for (int ni = 0; ni < 4; ++ni) {
      v16h b = *(const v16h*)(B0 + ((size_t)kb * 16 + ni) * 512);
      c[0][ni] = wmma16(a0, b, c[0][ni]);
      c[1][ni] = wmma16(a1, b, c[1][ni]);
    }
  }

  // 384 = 3*128: the whole block lives in one s -> scalar s/l, linear stores.
  const int s = bx / 3, lB = (bx % 3) * 128;
#pragma unroll
  for (int mi = 0; mi < 2; ++mi) {
    int l0 = lB + (mi2 * 2 + mi) * 16 + 8 * g;     // row = l0 + r, r<8
#pragma unroll
    for (int ni = 0; ni < 4; ++ni) {
      int nG = nBase + (ni2 * 4 + ni) * 16 + ln;
      int h = nG >> 5, d = nG & 31;
      int sh = s * HEADS + h;
      if (mode == 0) {
        h16* p = qsw + (size_t)(sh * 24 + (l0 >> 4)) * 512 +
                 ((l0 & 15) + 16 * (d >> 4)) * 16 + (d & 15);
#pragma unroll
        for (int r = 0; r < 8; ++r) p[r * 16] = (h16)(c[mi][ni][r] * QSC);
      } else if (mode == 1) {
        h16* p = ksw + (size_t)(sh * 24 + (l0 >> 4)) * 512 +
                 ((l0 & 15) + 16 * (d >> 4)) * 16 + (d & 15);
#pragma unroll
        for (int r = 0; r < 8; ++r) p[r * 16] = (h16)c[mi][ni][r];
      } else {
        h16* p = vsw + (size_t)((sh * 12 + (l0 >> 5)) * 2 + (d >> 4)) * 512 +
                 ((d & 15) + 16 * ((l0 >> 4) & 1)) * 16 + (l0 & 15);
#pragma unroll
        for (int r = 0; r < 8; ++r) p[r] = (h16)c[mi][ni][r];
      }
    }
  }
}

// ---------------------------------------------------------------------------
// Kernel 4: flash attention per (s,h,128 query rows). 8 waves, 16 rows each.
// K/V: double-buffered async-to-LDS staging (waves 0-3 copy K, 4-7 copy V);
// next block issued before computing current, drained with partial
// s_wait_asynccnt. P relaid C->A through per-wave LDS. Base-2 softmax.
// Output goes straight into the A-fragment layout of the output projection
// (k index = h*32+d, so k-block == head).
// ---------------------------------------------------------------------------
__global__ __launch_bounds__(256) void k_attn(
    const h16* __restrict__ qsw, const h16* __restrict__ ksw,
    const h16* __restrict__ vsw, h16* __restrict__ osw) {
  __shared__ __align__(32) h16 kF[2][4][32][16];  // double-buffered k^T tiles
  __shared__ __align__(32) h16 vF[2][4][32][16];  // double-buffered v tiles
  __shared__ __align__(64) h16 pB[8][16][32];     // per-wave P bounce
  const int lblk = blockIdx.x;   // 0..2, 128 query rows each
  const int h = blockIdx.y, s = blockIdx.z;
  const int t = threadIdx.x, lane = t & 31, wid = t >> 5;
  const int g = lane >> 4, ln = lane & 15;
  const int sh = s * HEADS + h;

  v16h qv = *(const v16h*)(qsw + (size_t)(sh * 24 + lblk * 8 + wid) * 512 + lane * 16);

  v8f zero = {0.f, 0.f, 0.f, 0.f, 0.f, 0.f, 0.f, 0.f};
  v8f o0 = zero, o1 = zero;
  float mi[8], li[8];
#pragma unroll
  for (int r = 0; r < 8; ++r) { mi[r] = -3.0e38f; li[r] = 0.f; }

  // Wave-uniform split: waves 0-3 stage K (4KB), waves 4-7 stage V (4KB).
  const int u = t & 127;
  const h16* sbase = (t < 128) ? (ksw + (size_t)sh * 24 * 512)
                               : (vsw + (size_t)sh * 24 * 512);
  h16* d0 = ((t < 128) ? &kF[0][0][0][0] : &vF[0][0][0][0]) + u * 16;
  h16* d1 = ((t < 128) ? &kF[1][0][0][0] : &vF[1][0][0][0]) + u * 16;

  cp_async32(sbase + u * 16, d0);                    // prologue: block 0
  for (int kb = 0; kb < 6; ++kb) {
    if (kb + 1 < 6) {
      cp_async32(sbase + (size_t)(kb + 1) * 2048 + u * 16, (kb & 1) ? d0 : d1);
      wait_async2();                                 // current buffer landed
    } else {
      wait_async0();
    }
    __syncthreads();
    const h16 (*kC)[32][16] = kF[kb & 1];
    const h16 (*vC)[32][16] = vF[kb & 1];
#pragma unroll
    for (int ms = 0; ms < 2; ++ms) {       // two 32-wide key sub-blocks
      v16h b0 = *(const v16h*)&kC[ms * 2 + 0][lane][0];
      v16h b1 = *(const v16h*)&kC[ms * 2 + 1][lane][0];
      v8f s0 = wmma16(qv, b0, zero);       // QSC (scale*log2e) folded into q
      v8f s1 = wmma16(qv, b1, zero);
#pragma unroll
      for (int r = 0; r < 8; ++r) {        // online softmax, row = r + 8*g
        float rm = redmax16(fmaxf(s0[r], s1[r]));
        float mn = fmaxf(mi[r], rm);
        float al = exp2_fast(mi[r] - mn);
        mi[r] = mn;
        float p0 = exp2_fast(s0[r] - mn);
        float p1 = exp2_fast(s1[r] - mn);
        li[r] = li[r] * al + redsum16(p0 + p1);
        o0[r] *= al;
        o1[r] *= al;
        int row = r + 8 * g;
        pB[wid][row][ln]      = (h16)p0;
        pB[wid][row][16 + ln] = (h16)p1;
      }
      Frag pf;                             // C-layout -> A-layout (in-wave DS)
      pf.u[0] = *(const uint4*)&pB[wid][ln][16 * g];
      pf.u[1] = *(const uint4*)&pB[wid][ln][16 * g + 8];
      o0 = wmma16(pf.v, *(const v16h*)&vC[ms * 2 + 0][lane][0], o0);
      o1 = wmma16(pf.v, *(const v16h*)&vC[ms * 2 + 1][lane][0], o1);
    }
    __syncthreads();
  }

  // Normalize, store into A-frag layout osw[mtile][kb=h][lane][e].
  int m0 = s * LEN + lblk * 128 + wid * 16 + 8 * g;   // + r
  h16* p = osw + ((size_t)(m0 >> 4) * 8 + h) * 512 +
           (8 * g + 16 * ((ln >> 3) & 1)) * 16 + (ln & 7);
#pragma unroll
  for (int r = 0; r < 8; ++r) {
    float inv = 1.0f / li[r];
    p[r * 16]     = (h16)(o0[r] * inv);    // d = ln      -> e = (ln&7)
    p[r * 16 + 8] = (h16)(o1[r] * inv);    // d = 16 + ln -> e = (ln&7)+8
  }
}

// ---------------------------------------------------------------------------
// Kernel 5: output projection, LDS-free GEMM, f32 row-major output.
// ---------------------------------------------------------------------------
__global__ __launch_bounds__(256) void k_oproj(
    const h16* __restrict__ osw, const h16* __restrict__ wsw,
    float* __restrict__ out) {
  const int bx = blockIdx.x, nBase = blockIdx.y * 128;
  const int t = threadIdx.x, lane = t & 31, wid = t >> 5;
  const int mi2 = wid >> 1, ni2 = wid & 1;
  const int g = lane >> 4, ln = lane & 15;
  const int mt0 = bx * 8 + mi2 * 2;
  const int nt0 = nBase / 16 + ni2 * 4;

  v8f zero = {0.f, 0.f, 0.f, 0.f, 0.f, 0.f, 0.f, 0.f};
  v8f c[2][4];
#pragma unroll
  for (int i = 0; i < 2; ++i)
#pragma unroll
    for (int j = 0; j < 4; ++j) c[i][j] = zero;

  const h16* A0 = osw + (size_t)mt0 * 8 * 512 + lane * 16;
  const h16* A1 = A0 + 8 * 512;
  const h16* B0 = wsw + ((size_t)3 * 8 * 16 + nt0) * 512 + lane * 16;  // Wo
#pragma unroll
  for (int kb = 0; kb < 8; ++kb) {
    v16h a0 = *(const v16h*)(A0 + kb * 512);
    v16h a1 = *(const v16h*)(A1 + kb * 512);
#pragma unroll
    for (int ni = 0; ni < 4; ++ni) {
      v16h b = *(const v16h*)(B0 + ((size_t)kb * 16 + ni) * 512);
      c[0][ni] = wmma16(a0, b, c[0][ni]);
      c[1][ni] = wmma16(a1, b, c[1][ni]);
    }
  }
#pragma unroll
  for (int mi = 0; mi < 2; ++mi) {
    int mG0 = bx * 128 + (mi2 * 2 + mi) * 16 + 8 * g;
#pragma unroll
    for (int ni = 0; ni < 4; ++ni) {
      int nG = nBase + (ni2 * 4 + ni) * 16 + ln;
      float* p = out + (size_t)mG0 * DIM + nG;
#pragma unroll
      for (int r = 0; r < 8; ++r) p[(size_t)r * DIM] = c[mi][ni][r];
    }
  }
}

// ---------------------------------------------------------------------------
extern "C" void kernel_launch(void* const* d_in, const int* in_sizes, int n_in,
                              void* d_out, int out_size, void* d_ws, size_t ws_size,
                              hipStream_t stream) {
  (void)in_sizes; (void)n_in; (void)out_size; (void)ws_size;
  const float* msa = (const float*)d_in[0];
  // d_in[1] = pair, d_in[6] = Wpb: dead code in the reference -> never read.
  const float* Wq = (const float*)d_in[2];
  const float* Wk = (const float*)d_in[3];
  const float* Wv = (const float*)d_in[4];
  const float* Wo = (const float*)d_in[5];
  float* out = (float*)d_out;

  char* ws = (char*)d_ws;
  const size_t MB = 1u << 20;
  const size_t SZ = (size_t)MTOT * DIM * sizeof(h16);   // 24 MB per tensor
  h16* wsw = (h16*)ws;                                  // 512 KB
  h16* asw = (h16*)(ws + MB);
  h16* qsw = (h16*)(ws + MB + SZ);
  h16* ksw = (h16*)(ws + MB + 2 * SZ);
  h16* vsw = (h16*)(ws + MB + 3 * SZ);
  h16* osw = (h16*)(ws + MB + 4 * SZ);

  k_pack_w<<<1024, 256, 0, stream>>>(Wq, Wk, Wv, Wo, wsw);
  k_swz_a<<<MTOT * 32 / 256, 256, 0, stream>>>(msa, asw);
  k_qkv<<<dim3(MTOT / 128, DIM / 128, 3), 256, 0, stream>>>(asw, wsw, qsw, ksw, vsw);
  k_attn<<<dim3(LEN / 128, HEADS, SEQ), 256, 0, stream>>>(qsw, ksw, vsw, osw);
  k_oproj<<<dim3(MTOT / 128, DIM / 128), 256, 0, stream>>>(osw, wsw, out);
}
